// Swin_46514495816319
// MI455X (gfx1250) — matrix operand order repaired and checked by reading, not compiled, not run on hardware
//
#include <hip/hip_runtime.h>
#include <hip/hip_bf16.h>
#include <math.h>
#include <stdint.h>

// ---------------------------------------------------------------------------
// Swin-T forward for MI455X (gfx1250, wave32, WMMA).
// v3: + async Global->LDS DMA (global_load_async_to_lds_b128 / s_wait_asynccnt)
// for the attention q/k staging; q-scale folded into softmax.
// GEMMs: fp16 operands (weights pre-converted once per launch, activations
// produced fp16 upstream), 2x2 register tiling -> 4 WMMAs / K-step, fp32 accum.
// Fragment layouts per CDNA5 ISA 7.12.2:
//   A 16x32 f16 : lanes 0-15 m=lane, elems[0..7]=K0..7, [8..15]=K16..23
//                 lanes 16-31 m=lane-16, elems[0..7]=K8..15, [8..15]=K24..31
//   B 32x16 f16 : lanes 0-15 n=lane, lanes 16-31 n=lane-16; elems hold the same K order as A
//   C/D f32     : reg r -> row (lane<16 ? r : 8+r), col = lane&15
// ---------------------------------------------------------------------------

typedef __attribute__((ext_vector_type(16))) _Float16 v16h;
typedef __attribute__((ext_vector_type(8)))  float    v8f;

__device__ __forceinline__ float gelu_exact(float x) {
    return 0.5f * x * (1.0f + erff(x * 0.70710678118654752440f));
}

// Async DMA: 16 bytes global -> LDS, tracked by ASYNCcnt (CDNA5 ISA).
// Generic LDS pointers carry the LDS byte offset in their low 32 bits.
__device__ __forceinline__ void async_g2l_b128(uint32_t lds_off, const _Float16* g) {
    asm volatile("global_load_async_to_lds_b128 %0, %1, off"
                 :: "v"(lds_off), "v"((uint64_t)(uintptr_t)g) : "memory");
}
__device__ __forceinline__ void wait_asynccnt0() {
    asm volatile("s_wait_asynccnt 0x0" ::: "memory");
}

__device__ __forceinline__ v16h frag_a16(const _Float16* __restrict__ row, int k0, int hi) {
    const _Float16* p = row + k0 + (hi ? 8 : 0);
    v16h a;
#pragma unroll
    for (int i = 0; i < 8; ++i) { a[i] = p[i]; a[8 + i] = p[16 + i]; }
    return a;
}

__device__ __forceinline__ v16h frag_b16(const _Float16* __restrict__ row, int k0, int hi) {
    const _Float16* p = row + k0 + (hi ? 8 : 0);
    v16h b;
#pragma unroll
    for (int i = 0; i < 8; ++i) { b[i] = p[i]; b[8 + i] = p[16 + i]; }
    return b;
}

// ---------------------------------------------------------------------------
// GEMM: Out[M,N] = act(X[M,K]h @ W[N,K]h^T + bias) (+res). fp32 accum.
// Wave computes a 32x32 block (2x2 tiles); WG (8 waves, 2x4) covers 64x128.
// M % 32 == 0, K % 32 == 0 at every call site; N ragged-guarded.
// ---------------------------------------------------------------------------
__global__ __launch_bounds__(256)
void gemm_wmma_h(const _Float16* __restrict__ X, const _Float16* __restrict__ Wt,
                 const float* __restrict__ bias, const float* __restrict__ res,
                 float* __restrict__ OutF, _Float16* __restrict__ OutH,
                 int M, int N, int K, int act) {
    const int lane = threadIdx.x & 31;
    const int wave = threadIdx.x >> 5;            // 0..7
    const int mw   = wave >> 2;                   // 0..1
    const int nwv  = wave & 3;                    // 0..3
    const int mbase = blockIdx.y * 64 + mw * 32;
    const int nbase = blockIdx.x * 128 + nwv * 32;
    if (mbase >= M || nbase >= N) return;         // wave-uniform exit

    const int lm = lane & 15;
    const int hi = lane >> 4;

    // clamped source rows (OOB lanes feed columns that are never stored)
    const int mr0 = mbase + lm;
    const int m1ok = (mbase + 16) < M;
    const int mr1 = m1ok ? (mbase + 16 + lm) : mr0;
    int nc0 = nbase + lm;        if (nc0 > N - 1) nc0 = N - 1;
    int nc1 = nbase + 16 + lm;   if (nc1 > N - 1) nc1 = N - 1;

    const _Float16* ar0 = X  + (size_t)mr0 * K;
    const _Float16* ar1 = X  + (size_t)mr1 * K;
    const _Float16* br0 = Wt + (size_t)nc0 * K;
    const _Float16* br1 = Wt + (size_t)nc1 * K;

    v8f c00 = {0.f,0.f,0.f,0.f,0.f,0.f,0.f,0.f};
    v8f c01 = c00, c10 = c00, c11 = c00;

    for (int k0 = 0; k0 < K; k0 += 32) {
        if (k0 + 64 < K) {
            __builtin_prefetch(ar0 + k0 + 64, 0, 1);
            __builtin_prefetch(br0 + k0 + 64, 0, 1);
        }
        const v16h a0 = frag_a16(ar0, k0, hi);
        const v16h a1 = frag_a16(ar1, k0, hi);
        const v16h b0 = frag_b16(br0, k0, hi);
        const v16h b1 = frag_b16(br1, k0, hi);
        c00 = __builtin_amdgcn_wmma_f32_16x16x32_f16(false, a0, false, b0, (short)0, c00, false, false);
        c01 = __builtin_amdgcn_wmma_f32_16x16x32_f16(false, a0, false, b1, (short)0, c01, false, false);
        c10 = __builtin_amdgcn_wmma_f32_16x16x32_f16(false, a1, false, b0, (short)0, c10, false, false);
        c11 = __builtin_amdgcn_wmma_f32_16x16x32_f16(false, a1, false, b1, (short)0, c11, false, false);
    }

    // ---- epilogue ----
    const int nA = nbase + lm;        // true (unclamped) columns
    const int nB = nbase + 16 + lm;
#pragma unroll
    for (int tm = 0; tm < 2; ++tm) {
        if (tm == 1 && !m1ok) break;
        const int rowb = mbase + tm * 16 + (hi ? 8 : 0);
#pragma unroll
        for (int tn = 0; tn < 2; ++tn) {
            const int n = (tn == 0) ? nA : nB;
            if (n >= N) continue;
            const v8f acc = (tm == 0) ? (tn == 0 ? c00 : c01) : (tn == 0 ? c10 : c11);
            const float bv = bias ? bias[n] : 0.0f;
#pragma unroll
            for (int r = 0; r < 8; ++r) {
                const int m = rowb + r;
                float v = acc[r] + bv;
                if (act) v = gelu_exact(v);
                if (res) v += res[(size_t)m * N + n];
                if (OutH) OutH[(size_t)m * N + n] = (_Float16)v;
                else      OutF[(size_t)m * N + n] = v;
            }
        }
    }
}

static void launch_gemm(const _Float16* X, const _Float16* Wt, const float* bias,
                        const float* res, float* OutF, _Float16* OutH,
                        int M, int N, int K, int act, hipStream_t s) {
    dim3 g((N + 127) / 128, (M + 63) / 64);
    gemm_wmma_h<<<g, 256, 0, s>>>(X, Wt, bias, res, OutF, OutH, M, N, K, act);
}

// ---------------------------------------------------------------------------
// Windowed attention: one WG (4 waves) per (window, head); fp16 in, fp16 out.
// q/k rows are DMA'd to LDS with global_load_async_to_lds_b128 (row-major,
// no transform); v goes through the VALU path because it needs a transpose.
// q-scale (hd^-0.5) is folded into the softmax pass.
// ---------------------------------------------------------------------------
__global__ __launch_bounds__(128)
void attn_win(const _Float16* __restrict__ qkv, const float* __restrict__ rpb,
              _Float16* __restrict__ out, int C, int heads, int nW, int nw,
              int H, int W, int shift) {
    const int wi = blockIdx.x / heads;       // global window index = b*nW + wl
    const int h  = blockIdx.x % heads;
    const int wl = wi % nW;
    const int wy = wl / nw, wx = wl % nw;
    const int threeC = 3 * C;
    const _Float16* base = qkv + (size_t)wi * 49 * threeC;
    const int qo = h * 32, ko = C + h * 32, vo = 2 * C + h * 32;

    __shared__ _Float16 qs[64 * 32];
    __shared__ _Float16 ks[64 * 32];
    __shared__ _Float16 vt[32 * 64];         // transposed: vt[d*64 + t]
    __shared__ float    S [64 * 65];
    __shared__ _Float16 Ps[64 * 64];

    const int tid = threadIdx.x;
    const float qscale = 0.17677669529663687f;  // 32^-0.5

    // v: regular path with transpose (DMA cannot transpose)
    for (int i = tid; i < 64 * 32; i += 128) {
        const int t = i >> 5, d = i & 31;
        float vv = 0.f;
        if (t < 49) vv = (float)base[(size_t)t * threeC + vo + d];
        vt[d * 64 + t] = (_Float16)vv;
    }
    // zero-fill qs/ks padding rows (disjoint from the DMA'd region)
    for (int i = tid; i < 15 * 32; i += 128) {
        const int t = 49 + (i >> 5), d = i & 31;
        qs[t * 32 + d] = (_Float16)0.0f;
        ks[t * 32 + d] = (_Float16)0.0f;
    }
    // async DMA q/k rows: 32 halfs = 4 x b128 per row; 16B-aligned for all C
    for (int i = tid; i < 49 * 4; i += 128) {
        const int t  = i >> 2;
        const int ch = (i & 3) * 8;
        const _Float16* grow = base + (size_t)t * threeC;
        async_g2l_b128((uint32_t)(uintptr_t)(qs + t * 32 + ch), grow + qo + ch);
        async_g2l_b128((uint32_t)(uintptr_t)(ks + t * 32 + ch), grow + ko + ch);
    }
    wait_asynccnt0();                        // this wave's DMAs done
    __syncthreads();                         // + everyone else's

    const int lane  = tid & 31;
    const int wave  = tid >> 5;              // rows [16w, 16w+16)
    const int mbase = wave * 16;
    const int lm    = lane & 15;
    const int hi    = lane >> 4;

    // ---- S = q @ k^T (unscaled) ----
    for (int j = 0; j < 4; ++j) {
        const v16h a = frag_a16(qs + (mbase + lm) * 32, 0, hi);
        const v16h b = frag_b16(ks + (j * 16 + lm) * 32, 0, hi);
        v8f acc = {0.f,0.f,0.f,0.f,0.f,0.f,0.f,0.f};
        acc = __builtin_amdgcn_wmma_f32_16x16x32_f16(false, a, false, b, (short)0, acc, false, false);
        const int row0 = mbase + (hi ? 8 : 0);
        const int col  = j * 16 + lm;
#pragma unroll
        for (int r = 0; r < 8; ++r) S[(row0 + r) * 65 + col] = acc[r];
    }
    __syncthreads();

    // ---- qscale + rpb + shift-mask + softmax (one thread per row) ----
    if (tid < 64) {
        const int row = tid;
        if (row < 49) {
            const int ry = row / 7, rx = row % 7;
            int idrow = 0;
            if (shift) {
                const int hr = wy * 7 + ry, wr = wx * 7 + rx;
                const int a1 = (hr < H - 7) ? 0 : ((hr < H - shift) ? 1 : 2);
                const int a2 = (wr < W - 7) ? 0 : ((wr < W - shift) ? 1 : 2);
                idrow = a1 * 3 + a2;
            }
            float mx = -1e30f;
            for (int c = 0; c < 49; ++c) {
                const int cy = c / 7, cx = c % 7;
                const int rel = (ry - cy + 6) * 13 + (rx - cx + 6);
                float v = S[row * 65 + c] * qscale + rpb[rel * heads + h];
                if (shift) {
                    const int hc = wy * 7 + cy, wc = wx * 7 + cx;
                    const int b1 = (hc < H - 7) ? 0 : ((hc < H - shift) ? 1 : 2);
                    const int b2 = (wc < W - 7) ? 0 : ((wc < W - shift) ? 1 : 2);
                    if (idrow != b1 * 3 + b2) v -= 100.0f;
                }
                S[row * 65 + c] = v;
                mx = fmaxf(mx, v);
            }
            float sum = 0.f;
            for (int c = 0; c < 49; ++c) {
                const float e = __expf(S[row * 65 + c] - mx);
                S[row * 65 + c] = e;
                sum += e;
            }
            const float inv = 1.0f / sum;
            for (int c = 0;  c < 49; ++c) Ps[row * 64 + c] = (_Float16)(S[row * 65 + c] * inv);
            for (int c = 49; c < 64; ++c) Ps[row * 64 + c] = (_Float16)0.0f;
        } else {
            for (int c = 0; c < 64; ++c) Ps[row * 64 + c] = (_Float16)0.0f;
        }
    }
    __syncthreads();

    // ---- O = P @ V ----
    for (int j = 0; j < 2; ++j) {
        v8f acc = {0.f,0.f,0.f,0.f,0.f,0.f,0.f,0.f};
        for (int k0 = 0; k0 < 64; k0 += 32) {
            const v16h a = frag_a16(Ps + (mbase + lm) * 64, k0, hi);
            const v16h b = frag_b16(vt + (j * 16 + lm) * 64, k0, hi);
            acc = __builtin_amdgcn_wmma_f32_16x16x32_f16(false, a, false, b, (short)0, acc, false, false);
        }
        const int row0 = mbase + (hi ? 8 : 0);
        const int col  = j * 16 + lm;
#pragma unroll
        for (int r = 0; r < 8; ++r) {
            const int m = row0 + r;
            if (m < 49) out[((size_t)wi * 49 + m) * C + h * 32 + col] = (_Float16)acc[r];
        }
    }
}

// ---------------------------------------------------------------------------
// Patch embed: 4x4/s4 conv + bias + LayerNorm. One block per token (fp32 out).
// ---------------------------------------------------------------------------
__global__ __launch_bounds__(128)
void patch_embed_ln(const float* __restrict__ x, const float* __restrict__ w,
                    const float* __restrict__ bias, const float* __restrict__ g,
                    const float* __restrict__ bb, float* __restrict__ out) {
    const int t  = blockIdx.x;
    const int b  = t / 3136;
    const int py = (t % 3136) / 56;
    const int px = t % 56;
    __shared__ float vals[96];
    __shared__ float red[128];
    const int tid = threadIdx.x;

    if (tid < 96) {
        float acc = bias[tid];
        const float* wb = w + tid * 48;
#pragma unroll
        for (int c = 0; c < 3; ++c)
#pragma unroll
            for (int i = 0; i < 4; ++i) {
                const float* xr = x + (((size_t)b * 3 + c) * 224 + (4 * py + i)) * 224 + 4 * px;
                const float* wr = wb + (c * 4 + i) * 4;
#pragma unroll
                for (int j = 0; j < 4; ++j) acc += xr[j] * wr[j];
            }
        vals[tid] = acc;
    }
    __syncthreads();

    red[tid] = (tid < 96) ? vals[tid] : 0.f; __syncthreads();
    for (int s = 64; s > 0; s >>= 1) { if (tid < s) red[tid] += red[tid + s]; __syncthreads(); }
    const float mean = red[0] / 96.f;
    __syncthreads();
    { float d = (tid < 96) ? (vals[tid] - mean) : 0.f; red[tid] = d * d; }
    __syncthreads();
    for (int s = 64; s > 0; s >>= 1) { if (tid < s) red[tid] += red[tid + s]; __syncthreads(); }
    const float rstd = rsqrtf(red[0] / 96.f + 1e-5f);
    if (tid < 96) out[(size_t)t * 96 + tid] = (vals[tid] - mean) * rstd * g[tid] + bb[tid];
}

// ---------------------------------------------------------------------------
// Row LayerNorm: fp32 in; fp16 out (yh) and/or fp32 out (yf), each nullable.
// ---------------------------------------------------------------------------
__global__ __launch_bounds__(256)
void ln_rows(const float* __restrict__ x, const float* __restrict__ g,
             const float* __restrict__ b, _Float16* __restrict__ yh,
             float* __restrict__ yf, int C) {
    __shared__ float red[256];
    const float* row = x + (size_t)blockIdx.x * C;
    _Float16* oh = yh ? (yh + (size_t)blockIdx.x * C) : (_Float16*)0;
    float*    of = yf ? (yf + (size_t)blockIdx.x * C) : (float*)0;
    const int tid = threadIdx.x;

    float s = 0.f;
    for (int c = tid; c < C; c += 256) s += row[c];
    red[tid] = s; __syncthreads();
    for (int st = 128; st > 0; st >>= 1) { if (tid < st) red[tid] += red[tid + st]; __syncthreads(); }
    const float mean = red[0] / C;
    __syncthreads();
    float v = 0.f;
    for (int c = tid; c < C; c += 256) { const float d = row[c] - mean; v += d * d; }
    red[tid] = v; __syncthreads();
    for (int st = 128; st > 0; st >>= 1) { if (tid < st) red[tid] += red[tid + st]; __syncthreads(); }
    const float rstd = rsqrtf(red[0] / C + 1e-5f);
    for (int c = tid; c < C; c += 256) {
        const float o = (row[c] - mean) * rstd * g[c] + b[c];
        if (oh) oh[c] = (_Float16)o;
        if (of) of[c] = o;
    }
}

// fp32 -> fp16 elementwise (weight pre-conversion)
__global__ void f32_to_f16(const float* __restrict__ in, _Float16* __restrict__ out, size_t n) {
    const size_t i = (size_t)blockIdx.x * blockDim.x + threadIdx.x;
    if (i < n) out[i] = (_Float16)in[i];
}

// Window partition gather (shift folded): fp16 [B,H,W,C] -> fp16 [B*nW*49,C]
__global__ void win_gather(const _Float16* __restrict__ xin, _Float16* __restrict__ xw,
                           int B, int H, int W, int C, int nw, int nW, int s) {
    const size_t idx = (size_t)blockIdx.x * blockDim.x + threadIdx.x;
    const size_t total = (size_t)B * nW * 49 * C;
    if (idx >= total) return;
    const int c  = (int)(idx % C);
    size_t t = idx / C;
    const int tt = (int)(t % 49);
    const size_t win = t / 49;
    const int b  = (int)(win / nW);
    const int wl = (int)(win % nW);
    const int wy = wl / nw, wx = wl % nw;
    const int r = tt / 7, cc = tt % 7;
    const int hh = (wy * 7 + r + s) % H;
    const int ww = (wx * 7 + cc + s) % W;
    xw[idx] = xin[(((size_t)b * H + hh) * W + ww) * C + c];
}

// Reverse window partition (+ reverse shift) with residual add into fp32 act.
__global__ void win_scatter_add(float* __restrict__ act, const float* __restrict__ y,
                                int B, int H, int W, int C, int nw, int nW, int s) {
    const size_t idx = (size_t)blockIdx.x * blockDim.x + threadIdx.x;
    const size_t total = (size_t)B * H * W * C;
    if (idx >= total) return;
    const int c = (int)(idx % C);
    size_t t = idx / C;
    const int ww = (int)(t % W); t /= W;
    const int hh = (int)(t % H);
    const int b  = (int)(t / H);
    int hr = hh - s; if (hr < 0) hr += H;
    int wr = ww - s; if (wr < 0) wr += W;
    const int wy = hr / 7, r = hr % 7, wx = wr / 7, cc = wr % 7;
    act[idx] += y[(((size_t)(b * nW + wy * nw + wx)) * 49 + r * 7 + cc) * C + c];
}

// Patch merge gather: fp32 [B,H,W,C] -> fp32 [B,H/2,W/2,4C], [x0,x1,x2,x3] blocks
__global__ void merge_gather(const float* __restrict__ xin, float* __restrict__ out,
                             int B, int H, int W, int C) {
    const int C4 = 4 * C, H2 = H / 2, W2 = W / 2;
    const size_t idx = (size_t)blockIdx.x * blockDim.x + threadIdx.x;
    const size_t total = (size_t)B * H2 * W2 * C4;
    if (idx >= total) return;
    const int c4 = (int)(idx % C4);
    size_t t = idx / C4;
    const int x2 = (int)(t % W2); t /= W2;
    const int y2 = (int)(t % H2);
    const int b  = (int)(t / H2);
    const int q = c4 / C, c = c4 % C;
    const int dy = q & 1;        // x0:(0,0) x1:(1,0) x2:(0,1) x3:(1,1)
    const int dx = q >> 1;
    out[idx] = xin[(((size_t)b * H + 2 * y2 + dy) * W + 2 * x2 + dx) * C + c];
}

// z[b,c,p] = hn[b,p,c]
__global__ void finalize_z(const float* __restrict__ hn, float* __restrict__ z,
                           int B, int C, int HW) {
    const size_t idx = (size_t)blockIdx.x * blockDim.x + threadIdx.x;
    const size_t total = (size_t)B * C * HW;
    if (idx >= total) return;
    const int p = (int)(idx % HW);
    size_t t = idx / HW;
    const int c = (int)(t % C);
    const int b = (int)(t / C);
    z[idx] = hn[((size_t)b * HW + p) * C + c];
}

__global__ void pool_mean(const float* __restrict__ hn, _Float16* __restrict__ pooled,
                          int B, int C, int HW) {
    const int i = blockIdx.x * blockDim.x + threadIdx.x;
    if (i >= B * C) return;
    const int b = i / C, c = i % C;
    float s = 0.f;
    for (int p = 0; p < HW; ++p) s += hn[((size_t)b * HW + p) * C + c];
    pooled[i] = (_Float16)(s / (float)HW);
}

// ---------------------------------------------------------------------------
// Host orchestration
// ---------------------------------------------------------------------------
extern "C" void kernel_launch(void* const* d_in, const int* in_sizes, int n_in,
                              void* d_out, int out_size, void* d_ws, size_t ws_size,
                              hipStream_t stream) {
    (void)in_sizes; (void)n_in; (void)out_size; (void)ws_size;
    int ip = 0;
    auto F = [&]() -> const float* { return (const float*)d_in[ip++]; };

    // --- unpack params in setup_inputs() insertion order (depth-first) ---
    const float* x        = F();
    const float* patch_w  = F();
    const float* patch_b  = F();
    const float* patch_ng = F();
    const float* patch_nb = F();

    struct BP { const float *n1g,*n1b,*qkvw,*qkvb,*projw,*projb,*rpb,*n2g,*n2b,
                            *fc1w,*fc1b,*fc2w,*fc2b; };
    struct MP { const float *ng,*nb,*redw; };
    static const int DEPTH_[4] = {2, 2, 6, 2};
    static const int HEADS_[4] = {3, 6, 12, 24};
    BP blocks[12]; MP merges[3];
    int bix = 0;
    for (int si = 0; si < 4; ++si) {
        for (int bi = 0; bi < DEPTH_[si]; ++bi) {
            BP& p = blocks[bix++];
            p.n1g = F(); p.n1b = F(); p.qkvw = F(); p.qkvb = F();
            p.projw = F(); p.projb = F(); p.rpb = F();
            p.n2g = F(); p.n2b = F();
            p.fc1w = F(); p.fc1b = F(); p.fc2w = F(); p.fc2b = F();
        }
        if (si < 3) { merges[si].ng = F(); merges[si].nb = F(); merges[si].redw = F(); }
    }
    const float* norm_g = F();
    const float* norm_b = F();
    const float* head_w = F();
    const float* head_b = F();

    // --- workspace carve: fp32 region then fp16 region (~340 MB peak) ---
    float* A  = (float*)d_ws;                         // residual stream [B,H,W,C] fp32
    float* Pb = A + ((size_t)10 << 20);               // proj-out / merge-gather / final-LN fp32
    _Float16* hb   = (_Float16*)(Pb + ((size_t)10 << 20));
    _Float16* T1h  = hb;                              // LN out (fp16)          10M
    _Float16* XWh  = T1h  + ((size_t)10 << 20);       // windowed x             10M
    _Float16* QKVh = XWh  + ((size_t)10 << 20);       // qkv                    30M
    _Float16* Ybh  = QKVh + ((size_t)30 << 20);       // attention out          10M
    _Float16* Hmh  = Ybh  + ((size_t)10 << 20);       // MLP hidden             40M
    _Float16* PLh  = Hmh  + ((size_t)40 << 20);       // pooled [32,768]
    _Float16* WH   = PLh  + 32768;                    // converted weights (~28M)

    size_t wcur = 0;
    auto CV = [&](const float* src, size_t n) -> const _Float16* {
        _Float16* dst = WH + wcur;
        wcur += (n + 63) & ~(size_t)63;               // keep 128B alignment
        f32_to_f16<<<(unsigned)((n + 255) / 256), 256, 0, stream>>>(src, dst, n);
        return dst;
    };

    // pre-convert all GEMM weights to fp16 (one pass over ~28M params)
    struct BPH { const _Float16 *qkv, *proj, *fc1, *fc2; };
    BPH bh[12]; const _Float16* redh[3]; const _Float16* headh;
    {
        int C = 96; bix = 0;
        for (int si = 0; si < 4; ++si) {
            for (int bi = 0; bi < DEPTH_[si]; ++bi) {
                const BP& p = blocks[bix];
                bh[bix].qkv  = CV(p.qkvw, (size_t)3 * C * C);
                bh[bix].proj = CV(p.projw, (size_t)C * C);
                bh[bix].fc1  = CV(p.fc1w, (size_t)4 * C * C);
                bh[bix].fc2  = CV(p.fc2w, (size_t)4 * C * C);
                ++bix;
            }
            if (si < 3) { redh[si] = CV(merges[si].redw, (size_t)8 * C * C); C *= 2; }
        }
        headh = CV(head_w, (size_t)1000 * 768);
    }

    const int B = 32;
    patch_embed_ln<<<B * 56 * 56, 128, 0, stream>>>(x, patch_w, patch_b,
                                                    patch_ng, patch_nb, A);

    int Hc = 56, Wc = 56, C = 96;
    bix = 0;
    for (int si = 0; si < 4; ++si) {
        const int heads = HEADS_[si];
        const int nh = Hc / 7, nw = Wc / 7, nW = nh * nw;
        const int rows = B * Hc * Wc;                 // token count (== B*nW*49)
        const size_t tot = (size_t)rows * C;

        for (int bi = 0; bi < DEPTH_[si]; ++bi) {
            const BP& p = blocks[bix];
            const BPH& ph = bh[bix];
            ++bix;
            int shift = (bi & 1) ? 3 : 0;
            if (7 >= ((Hc < Wc) ? Hc : Wc)) shift = 0;

            ln_rows<<<rows, 256, 0, stream>>>(A, p.n1g, p.n1b, T1h, (float*)0, C);
            win_gather<<<(unsigned)((tot + 255) / 256), 256, 0, stream>>>(
                T1h, XWh, B, Hc, Wc, C, nw, nW, shift);
            launch_gemm(XWh, ph.qkv, p.qkvb, nullptr, nullptr, QKVh,
                        rows, 3 * C, C, 0, stream);
            attn_win<<<B * nW * heads, 128, 0, stream>>>(
                QKVh, p.rpb, Ybh, C, heads, nW, nw, Hc, Wc, shift);
            launch_gemm(Ybh, ph.proj, p.projb, nullptr, Pb, nullptr,
                        rows, C, C, 0, stream);
            win_scatter_add<<<(unsigned)((tot + 255) / 256), 256, 0, stream>>>(
                A, Pb, B, Hc, Wc, C, nw, nW, shift);

            ln_rows<<<rows, 256, 0, stream>>>(A, p.n2g, p.n2b, T1h, (float*)0, C);
            launch_gemm(T1h, ph.fc1, p.fc1b, nullptr, nullptr, Hmh,
                        rows, 4 * C, C, /*gelu*/1, stream);
            launch_gemm(Hmh, ph.fc2, p.fc2b, /*res*/A, A, nullptr,
                        rows, C, 4 * C, 0, stream);
        }

        if (si < 3) {
            const int H2 = Hc / 2, W2 = Wc / 2;
            const size_t mt = (size_t)B * H2 * W2 * 4 * C;
            merge_gather<<<(unsigned)((mt + 255) / 256), 256, 0, stream>>>(A, Pb, B, Hc, Wc, C);
            ln_rows<<<B * H2 * W2, 256, 0, stream>>>(Pb, merges[si].ng, merges[si].nb,
                                                     T1h, (float*)0, 4 * C);
            launch_gemm(T1h, redh[si], nullptr, nullptr, A, nullptr,
                        B * H2 * W2, 2 * C, 4 * C, 0, stream);
            Hc = H2; Wc = W2; C *= 2;
        }
    }

    // --- final: Hc=Wc=7, C=768. out = [logits(32x1000) | z(32x768x7x7)] ---
    float* out = (float*)d_out;
    const int HW = Hc * Wc;                           // 49
    ln_rows<<<B * HW, 256, 0, stream>>>(A, norm_g, norm_b, (_Float16*)0, Pb, C);
    finalize_z<<<(unsigned)(((size_t)B * C * HW + 255) / 256), 256, 0, stream>>>(
        Pb, out + (size_t)B * 1000, B, C, HW);
    pool_mean<<<(B * C + 255) / 256, 256, 0, stream>>>(Pb, PLh, B, C, HW);
    launch_gemm(PLh, headh, head_b, nullptr, out, nullptr, B, 1000, C, 0, stream);
}
